// Sinkhorn_Net_48799418417681
// MI455X (gfx1250) — matrix operand order repaired and checked by reading, not compiled, not run on hardware
//
#include <hip/hip_runtime.h>

typedef __attribute__((ext_vector_type(16))) _Float16 v16h;
typedef __attribute__((ext_vector_type(8)))  float    v8f;
typedef int i32x4 __attribute__((vector_size(16)));   // matches async-LDS builtin param

#define N_DIM   128
#define BATCH   4096
#define TSTRIDE 129   // padded LDS row stride: (129*i + n) % 64 conflict-free both ways
#define N_ITER  20

#define ASG __attribute__((address_space(1)))
#define ASL __attribute__((address_space(3)))

#if defined(__gfx1250__) && __has_builtin(__builtin_amdgcn_global_load_async_to_lds_b128) && \
    __has_builtin(__builtin_amdgcn_s_wait_asynccnt)
#define HAVE_ASYNC_LDS 1
#else
#define HAVE_ASYNC_LDS 0
#endif

__device__ __forceinline__ float wave_max32(float v) {
#pragma unroll
  for (int m = 16; m >= 1; m >>= 1) v = fmaxf(v, __shfl_xor(v, m, 32));
  return v;
}
__device__ __forceinline__ float wave_sum32(float v) {
#pragma unroll
  for (int m = 16; m >= 1; m >>= 1) v += __shfl_xor(v, m, 32);
  return v;
}

// ---------------------------------------------------------------------------
// Prep: swizzle W1/b1 (f32) and W2 (->f16) into per-lane WMMA fragment order.
//  A-frag (16x32 f16, ISA 7.12.2): lane m=lane%16 holds row M=m; element j maps
//    to K = (j<8 ? j : j+8) + 8*(lane>=16).
//  B-frag (32x16 f16): lane holds col N=lane%16; element j -> K = (j<8 ? j : j+8) + 8*(lane>=16).
// W2h layout: [kc(32)][ni(8)][lane(32)][j(16)] halves  -> 32B contiguous per lane
// W1s/b1s layout: [kc(32)][lane(32)][j(16)] floats     -> 64B contiguous per lane
// ---------------------------------------------------------------------------
__global__ __launch_bounds__(256)
void prep_swizzle(const float* __restrict__ W1, const float* __restrict__ b1,
                  const float* __restrict__ W2,
                  _Float16* __restrict__ W2h, float* __restrict__ W1s,
                  float* __restrict__ b1s) {
  int t = blockIdx.x * 256 + threadIdx.x;
  if (t < 32 * 8 * 32 * 16) {            // 131072 entries of W2h
    int j    = t & 15;
    int lane = (t >> 4) & 31;
    int cn   = t >> 9;                   // kc*8 + ni
    int ni   = cn & 7;
    int kc   = cn >> 3;
    int k = kc * 32 + (j < 8 ? j : j + 8) + 8 * (lane >> 4);
    int n = ni * 16 + (lane & 15);
    W2h[t] = (_Float16)W2[k * N_DIM + n];
  }
  int u = t - 32 * 8 * 32 * 16;
  if (u >= 0 && u < 32 * 32 * 16) {      // 16384 entries of W1s/b1s
    int j    = u & 15;
    int lane = (u >> 4) & 31;
    int kc   = u >> 9;
    int k = kc * 32 + (j < 8 ? j : j + 8) + 8 * (lane >> 4);
    W1s[u] = W1[k];
    b1s[u] = b1[k];
  }
}

// ---------------------------------------------------------------------------
// Main: one block (8 wave32) per batch element.
//   Phase 0: async-stream gumbel tile HBM -> LDS (overlaps with Phase 1)
//   Phase 1: log_alpha = relu(x*W1+b1) @ W2         (f16 WMMA, f32 accum)
//   Spill  : la(=gumbel) += acc + b2   (TEMP=1, NOISE_FACTOR=1)
//   Phase 3: 20 Sinkhorn row/col log-normalizations, fully in LDS
//   Phase 4: inv_perms = exp(la)^T to HBM; out = inv_perms @ x via wave reduce
// ---------------------------------------------------------------------------
__global__ __launch_bounds__(256)
void sinkhorn_main(const float* __restrict__ x,
                   const float* __restrict__ b2,
                   const float* __restrict__ gumbel,
                   const float* __restrict__ W1s,
                   const float* __restrict__ b1s,
                   const _Float16* __restrict__ W2h,
                   float* __restrict__ out_perm,
                   float* __restrict__ out_vec) {
  extern __shared__ float smem[];
  float* la  = smem;                       // 128 x 129
  float* xs  = smem + N_DIM * TSTRIDE;     // 128
  float* b2s = xs + N_DIM;                 // 128

  const int b    = blockIdx.x;
  const int tid  = threadIdx.x;
  const int lane = tid & 31;
  const int wv   = tid >> 5;               // wave id 0..7 -> row-tile mi

  const float* gbase = gumbel + (size_t)b * N_DIM * N_DIM;

#if HAVE_ASYNC_LDS
  // ---------------- Phase 0: async gumbel -> LDS (la), padded layout --------
  // 16 instructions/wave, 512B coalesced global per instruction; ASYNCcnt<=16.
#pragma unroll
  for (int c = 0; c < 16; ++c) {
    int idx4 = c * 256 + tid;
    int flat = idx4 * 4;
    int i = flat >> 7, n = flat & 127;
    ASG i32x4* g = (ASG i32x4*)(unsigned long long)(size_t)(gbase + flat);
    ASL i32x4* l = (ASL i32x4*)(unsigned int)(size_t)(la + i * TSTRIDE + n);
    __builtin_amdgcn_global_load_async_to_lds_b128(g, l, 0, 0);
  }
#else
  __builtin_prefetch(gbase + tid * 64, 0, 0);   // global_prefetch_b8 warm-up
#endif

  if (tid < N_DIM) { xs[tid] = x[b * N_DIM + tid]; b2s[tid] = b2[tid]; }
  __syncthreads();

  // ---------------- Phase 1: MLP via WMMA ----------------
  const float xv = xs[wv * 16 + (lane & 15)];   // row scalar for this lane's A rows
  v8f acc[8] = {};

  for (int kc = 0; kc < 32; ++kc) {
    // Build A fragment: h = relu(x * W1 + b1) in f16, per-lane contiguous loads
    const float4* w1p = (const float4*)(W1s + (size_t)(kc * 32 + lane) * 16);
    const float4* b1p = (const float4*)(b1s + (size_t)(kc * 32 + lane) * 16);
    float hw[16], hb[16];
    ((float4*)hw)[0] = w1p[0]; ((float4*)hw)[1] = w1p[1];
    ((float4*)hw)[2] = w1p[2]; ((float4*)hw)[3] = w1p[3];
    ((float4*)hb)[0] = b1p[0]; ((float4*)hb)[1] = b1p[1];
    ((float4*)hb)[2] = b1p[2]; ((float4*)hb)[3] = b1p[3];
    v16h afrag;
#pragma unroll
    for (int j = 0; j < 16; ++j)
      afrag[j] = (_Float16)fmaxf(fmaf(xv, hw[j], hb[j]), 0.0f);

#pragma unroll
    for (int ni = 0; ni < 8; ++ni) {
      v16h bfrag = *(const v16h*)(W2h + ((size_t)((kc * 8 + ni) * 32 + lane)) * 16);
      acc[ni] = __builtin_amdgcn_wmma_f32_16x16x32_f16(
          false, afrag, false, bfrag, (short)0, acc[ni], false, false);
    }
  }

  // ---------------- Spill accumulators into LDS tile ----------------
#if HAVE_ASYNC_LDS
  __builtin_amdgcn_s_wait_asynccnt(0);   // gumbel tile fully landed (this wave)
  __syncthreads();                       // ... and for all waves
  {
    const int halfsel = lane >> 4;
    const int ncol0   = lane & 15;
#pragma unroll
    for (int ni = 0; ni < 8; ++ni) {
      const int n   = ni * 16 + ncol0;
      const float c = b2s[n];
#pragma unroll
      for (int r = 0; r < 8; ++r) {
        const int i = wv * 16 + r + 8 * halfsel;
        la[i * TSTRIDE + n] += acc[ni][r] + c;   // la held gumbel
      }
    }
  }
  __syncthreads();
#else
  {
    const int halfsel = lane >> 4;
    const int ncol0   = lane & 15;
#pragma unroll
    for (int ni = 0; ni < 8; ++ni) {
      const int n   = ni * 16 + ncol0;
      const float c = b2s[n];
#pragma unroll
      for (int r = 0; r < 8; ++r) {
        const int i = wv * 16 + r + 8 * halfsel;
        la[i * TSTRIDE + n] = acc[ni][r] + c;
      }
    }
  }
  __syncthreads();
  // gumbel add (coalesced float4 through VGPRs)
  {
    const float4* gp = (const float4*)gbase;
#pragma unroll 4
    for (int c = 0; c < 16; ++c) {
      int idx4 = c * 256 + tid;
      float4 g = gp[idx4];
      int flat = idx4 * 4;
      int i = flat >> 7, n = flat & 127;
      float* p = &la[i * TSTRIDE + n];
      p[0] += g.x; p[1] += g.y; p[2] += g.z; p[3] += g.w;
    }
  }
  __syncthreads();
#endif

  // ---------------- Phase 3: Sinkhorn in LDS ----------------
  for (int it = 0; it < N_ITER; ++it) {
    // Row log-normalization: wave wv handles rows wv*16..wv*16+15
#pragma unroll 1
    for (int q = 0; q < 16; ++q) {
      const int i = wv * 16 + q;
      float v0 = la[i * TSTRIDE + lane];
      float v1 = la[i * TSTRIDE + lane + 32];
      float v2 = la[i * TSTRIDE + lane + 64];
      float v3 = la[i * TSTRIDE + lane + 96];
      float m = wave_max32(fmaxf(fmaxf(v0, v1), fmaxf(v2, v3)));
      float s = __expf(v0 - m) + __expf(v1 - m) + __expf(v2 - m) + __expf(v3 - m);
      s = wave_sum32(s);
      float lse = m + __logf(s);
      la[i * TSTRIDE + lane]      = v0 - lse;
      la[i * TSTRIDE + lane + 32] = v1 - lse;
      la[i * TSTRIDE + lane + 64] = v2 - lse;
      la[i * TSTRIDE + lane + 96] = v3 - lse;
    }
    __syncthreads();
    // Column log-normalization: wave wv handles cols wv*16..wv*16+15
#pragma unroll 1
    for (int q = 0; q < 16; ++q) {
      const int n = wv * 16 + q;
      float v0 = la[(lane)      * TSTRIDE + n];
      float v1 = la[(lane + 32) * TSTRIDE + n];
      float v2 = la[(lane + 64) * TSTRIDE + n];
      float v3 = la[(lane + 96) * TSTRIDE + n];
      float m = wave_max32(fmaxf(fmaxf(v0, v1), fmaxf(v2, v3)));
      float s = __expf(v0 - m) + __expf(v1 - m) + __expf(v2 - m) + __expf(v3 - m);
      s = wave_sum32(s);
      float lse = m + __logf(s);
      la[(lane)      * TSTRIDE + n] = v0 - lse;
      la[(lane + 32) * TSTRIDE + n] = v1 - lse;
      la[(lane + 64) * TSTRIDE + n] = v2 - lse;
      la[(lane + 96) * TSTRIDE + n] = v3 - lse;
    }
    __syncthreads();
  }

  // ---------------- Phase 4: outputs ----------------
  // inv_perms[b,i,j] = exp(la[j,i])  (transposed, coalesced stores)
  {
    float* op = out_perm + (size_t)b * N_DIM * N_DIM;
#pragma unroll 2
    for (int c = 0; c < 64; ++c) {
      int flat = c * 256 + tid;
      int i = flat >> 7, j = flat & 127;
      op[flat] = __expf(la[j * TSTRIDE + i]);
    }
    // out[b,i] = sum_j exp(la[j,i]) * x[j]
#pragma unroll 1
    for (int q = 0; q < 16; ++q) {
      int i = wv * 16 + q;
      float t = 0.0f;
#pragma unroll
      for (int c = 0; c < 4; ++c) {
        int j = lane + 32 * c;
        t = fmaf(__expf(la[j * TSTRIDE + i]), xs[j], t);
      }
      t = wave_sum32(t);
      if (lane == 0) out_vec[(size_t)b * N_DIM + i] = t;
    }
  }
}

extern "C" void kernel_launch(void* const* d_in, const int* in_sizes, int n_in,
                              void* d_out, int out_size, void* d_ws, size_t ws_size,
                              hipStream_t stream) {
  const float* x      = (const float*)d_in[0];
  const float* W1     = (const float*)d_in[1];
  const float* b1     = (const float*)d_in[2];
  const float* W2     = (const float*)d_in[3];
  const float* b2     = (const float*)d_in[4];
  const float* gumbel = (const float*)d_in[5];
  float* out = (float*)d_out;

  // Workspace carve-up: W2h (256KB f16, swizzled) | W1s (64KB) | b1s (64KB)
  char* ws = (char*)d_ws;
  _Float16* W2h = (_Float16*)ws;
  float*    W1s = (float*)(ws + 262144);
  float*    b1s = (float*)(ws + 262144 + 65536);

  // 131072 + 16384 = 147456 work items -> 576 blocks of 256
  prep_swizzle<<<576, 256, 0, stream>>>(W1, b1, W2, W2h, W1s, b1s);

  size_t shmem = (size_t)(N_DIM * TSTRIDE + 2 * N_DIM) * sizeof(float); // 67072 B
  sinkhorn_main<<<BATCH, 256, shmem, stream>>>(
      x, b2, gumbel, W1s, b1s, W2h,
      out, out + (size_t)BATCH * N_DIM * N_DIM);
}
